// PC_Trs_30331059045147
// MI455X (gfx1250) — matrix operand order repaired and checked by reading, not compiled, not run on hardware
//
#include <hip/hip_runtime.h>
#include <hip/hip_bf16.h>
#include <math.h>
#include <stdint.h>

// ---- problem constants (match reference) ----
#define PN     4096
#define NB     4
#define CIN    3
#define CKP    61
#define KP     15
#define FF     256
#define NH     8
#define NL     4
#define HIDN   64
#define NS     32
#define RADIUS 0.12f
#define SIGK   0.06f
#define WINW   0.25f
#define GG     4
#define NCELL  64
#define MCAP   160
#define NBINS  5
#define DD     64      // CKP + 3

typedef _Float16 v16h __attribute__((ext_vector_type(16)));
typedef float    v8f  __attribute__((ext_vector_type(8)));

// ============================================================
// 1) Ball query (top-32 within radius) fused with KPConv
// ============================================================
__global__ void kpconv_kernel(const float* __restrict__ pos,
                              const float* __restrict__ fin,
                              const float* __restrict__ kp,
                              const float* __restrict__ kpW,
                              float* __restrict__ fout) {
  __shared__ float s_kp[KP * 3];
  __shared__ float s_W[KP * CIN * CKP];
  for (int i = threadIdx.x; i < KP * 3; i += blockDim.x) s_kp[i] = kp[i];
  for (int i = threadIdx.x; i < KP * CIN * CKP; i += blockDim.x) s_W[i] = kpW[i];
  __syncthreads();

  int gid = blockIdx.x * blockDim.x + threadIdx.x;
  if (gid >= NB * PN) return;
  int b = gid / PN, p = gid % PN;
  const float* pb = pos + (size_t)b * PN * 3;
  float px = pb[p * 3 + 0], py = pb[p * 3 + 1], pz = pb[p * 3 + 2];

  float nd[NS];
  int   ni[NS];
  int cnt = 0;
  const float R2 = RADIUS * RADIUS;
  for (int j = 0; j < PN; ++j) {
    float dx = pb[j * 3 + 0] - px;
    float dy = pb[j * 3 + 1] - py;
    float dz = pb[j * 3 + 2] - pz;
    float d2 = dx * dx + dy * dy + dz * dz;
    if (d2 > R2) continue;
    if (cnt < NS) {
      int q = cnt++;
      while (q > 0 && nd[q - 1] > d2) { nd[q] = nd[q - 1]; ni[q] = ni[q - 1]; --q; }
      nd[q] = d2; ni[q] = j;
    } else if (d2 < nd[NS - 1]) {
      int q = NS - 1;
      while (q > 0 && nd[q - 1] > d2) { nd[q] = nd[q - 1]; ni[q] = ni[q - 1]; --q; }
      nd[q] = d2; ni[q] = j;
    }
  }

  float fk[KP * CIN];
  for (int i = 0; i < KP * CIN; ++i) fk[i] = 0.f;
  const float* fb = fin + (size_t)b * PN * CIN;
  for (int s = 0; s < cnt; ++s) {
    int j = ni[s];
    float rx = pb[j * 3 + 0] - px;
    float ry = pb[j * 3 + 1] - py;
    float rz = pb[j * 3 + 2] - pz;
    float f0 = fb[j * 3 + 0], f1 = fb[j * 3 + 1], f2 = fb[j * 3 + 2];
    for (int kk = 0; kk < KP; ++kk) {
      float ax = rx - s_kp[kk * 3 + 0];
      float ay = ry - s_kp[kk * 3 + 1];
      float az = rz - s_kp[kk * 3 + 2];
      float dk = sqrtf(ax * ax + ay * ay + az * az);
      float infl = 1.0f - dk * (1.0f / SIGK);
      infl = infl > 0.f ? infl : 0.f;
      fk[kk * 3 + 0] += infl * f0;
      fk[kk * 3 + 1] += infl * f1;
      fk[kk * 3 + 2] += infl * f2;
    }
  }
  float* op = fout + (size_t)gid * CKP;
  for (int d = 0; d < CKP; ++d) {
    float acc = 0.f;
    for (int i = 0; i < KP * CIN; ++i) acc += fk[i] * s_W[i * CKP + d];
    op[d] = acc;
  }
}

// ============================================================
// 2) BatchNorm stats + apply + leaky relu
// ============================================================
__global__ void bn_stats_kernel(const float* __restrict__ feat, float* __restrict__ stats) {
  int c = blockIdx.x;
  int t = threadIdx.x;
  float s = 0.f, s2 = 0.f;
  for (int i = t; i < NB * PN; i += 256) {
    float v = feat[(size_t)i * CKP + c];
    s += v; s2 += v * v;
  }
  __shared__ float rs[256], rq[256];
  rs[t] = s; rq[t] = s2;
  __syncthreads();
  for (int k = 128; k > 0; k >>= 1) {
    if (t < k) { rs[t] += rs[t + k]; rq[t] += rq[t + k]; }
    __syncthreads();
  }
  if (t == 0) {
    float mean = rs[0] / (float)(NB * PN);
    float var  = rq[0] / (float)(NB * PN) - mean * mean;
    stats[c] = mean;
    stats[CKP + c] = var;
  }
}

__global__ void bn_apply_kernel(float* __restrict__ feat, const float* __restrict__ stats,
                                const float* __restrict__ gamma, const float* __restrict__ beta) {
  int i = blockIdx.x * blockDim.x + threadIdx.x;
  if (i >= NB * PN * CKP) return;
  int c = i % CKP;
  float y = (feat[i] - stats[c]) * rsqrtf(stats[CKP + c] + 1e-5f) * gamma[c] + beta[c];
  feat[i] = y >= 0.f ? y : 0.2f * y;
}

// ============================================================
// 3) Cell ids + deterministic per-cell adaptive max pool
// ============================================================
__global__ void cellid_kernel(const float* __restrict__ pos, int* __restrict__ cellid) {
  int i = blockIdx.x * blockDim.x + threadIdx.x;
  if (i >= NB * PN) return;
  int cc[3];
  for (int a = 0; a < 3; ++a) {
    int ci = (int)floorf(pos[(size_t)i * 3 + a] / WINW);
    ci = ci < 0 ? 0 : (ci > GG - 1 ? GG - 1 : ci);
    cc[a] = ci;
  }
  cellid[i] = cc[0] * GG * GG + cc[1] * GG + cc[2];
}

__global__ void cluster_pool_kernel(const float* __restrict__ pos,
                                    const float* __restrict__ feat,
                                    const int* __restrict__ cellid,
                                    float* __restrict__ Aemb) {
  int blk = blockIdx.x;
  int cell = blk % NCELL, b = blk / NCELL;
  int d = threadIdx.x;  // 0..63
  const int* cid = cellid + (size_t)b * PN;

  int cnt = 0;
  for (int p = 0; p < PN; ++p) cnt += (cid[p] == cell) ? 1 : 0;
  int n = cnt < MCAP ? cnt : MCAP;

  int bs[NBINS], be[NBINS];
  for (int i = 0; i < NBINS; ++i) {
    bs[i] = (i * n) / NBINS;
    be[i] = ((i + 1) * n + NBINS - 1) / NBINS;
  }
  float mx[NBINS];
  for (int i = 0; i < NBINS; ++i) mx[i] = -3.4e38f;

  int r = 0;
  for (int p = 0; p < PN && r < MCAP; ++p) {
    if (cid[p] != cell) continue;
    float val = (d < 3) ? pos[((size_t)b * PN + p) * 3 + d]
                        : feat[((size_t)b * PN + p) * CKP + (d - 3)];
    for (int i = 0; i < NBINS; ++i)
      if (r >= bs[i] && r < be[i]) mx[i] = fmaxf(mx[i], val);
    ++r;
  }
  for (int i = 0; i < NBINS; ++i) {
    int tok = (cell * NBINS + i) * NB + b;
    Aemb[(size_t)tok * DD + d] = (n > 0) ? mx[i] : 0.0f;
  }
}

// ============================================================
// 4) WMMA GEMM, templated epilogue + compile-time K (full unroll).
//    B tile (32x16 f32, shared by all 4 waves) staged into LDS via
//    global_load_async_to_lds_b128 (ASYNCcnt), A via b128 loads.
// ============================================================
template <int KDIM, bool HAS_BIAS, bool DO_RELU, bool HAS_RESID>
__global__ __launch_bounds__(128) void gemm_wmma_kernel(
    const float* __restrict__ A, const float* __restrict__ Bm,
    const float* __restrict__ bias, const float* __restrict__ resid,
    float* __restrict__ C, int Mdim, int Ndim) {
  __shared__ __align__(16) float sB[32 * 16];

  const int t = threadIdx.x;
  const int lane = t & 31;
  const int wave = t >> 5;
  const int lo = lane & 15;
  const int hi = (lane >> 4) & 1;
  const int rowBase = (blockIdx.y * 4 + wave) * 16;
  const int colBase = blockIdx.x * 16;

  const float* Arow = A + (size_t)(rowBase + lo) * KDIM;
  v8f acc = {0.f, 0.f, 0.f, 0.f, 0.f, 0.f, 0.f, 0.f};

  // each of 128 threads copies 16B of the B tile: thread t -> row t/4, cols (t%4)*4..+3
  const unsigned ldsDst = (unsigned)(uintptr_t)&sB[0] + (unsigned)(t * 16);
  const int brow  = t >> 2;
  const int bcol4 = (t & 3) * 4;

#pragma unroll
  for (int k0 = 0; k0 < KDIM; k0 += 32) {
    __syncthreads();  // previous iteration's LDS reads complete
    {
      unsigned long long gaddr = (unsigned long long)(uintptr_t)
          (Bm + (size_t)(k0 + brow) * Ndim + colBase + bcol4);
      asm volatile("global_load_async_to_lds_b128 %0, %1, off"
                   :: "v"(ldsDst), "v"(gaddr) : "memory");
      asm volatile("s_wait_asynccnt 0x0" ::: "memory");
    }
    __syncthreads();  // tile visible to all waves

    // A 16x32 f16 fragment (ISA 7.12.2): lane<16 -> K {0..7,16..23}, lane>=16 -> {8..15,24..31}
    v16h afrag;
    {
      const float4* Ap  = reinterpret_cast<const float4*>(Arow + k0 + hi * 8);
      const float4* Ap2 = reinterpret_cast<const float4*>(Arow + k0 + 16 + hi * 8);
      float4 a0 = Ap[0], a1 = Ap[1], a2 = Ap2[0], a3 = Ap2[1];
      afrag[0]  = (_Float16)a0.x; afrag[1]  = (_Float16)a0.y;
      afrag[2]  = (_Float16)a0.z; afrag[3]  = (_Float16)a0.w;
      afrag[4]  = (_Float16)a1.x; afrag[5]  = (_Float16)a1.y;
      afrag[6]  = (_Float16)a1.z; afrag[7]  = (_Float16)a1.w;
      afrag[8]  = (_Float16)a2.x; afrag[9]  = (_Float16)a2.y;
      afrag[10] = (_Float16)a2.z; afrag[11] = (_Float16)a2.w;
      afrag[12] = (_Float16)a3.x; afrag[13] = (_Float16)a3.y;
      afrag[14] = (_Float16)a3.z; afrag[15] = (_Float16)a3.w;
    }
    // B 32x16 f16 fragment from LDS: N striped across lanes, K in same order as A
    v16h bfrag;
#pragma unroll
    for (int e = 0; e < 16; ++e)
      bfrag[e] = (_Float16)sB[((e & 8) * 2 + hi * 8 + (e & 7)) * 16 + lo];

    acc = __builtin_amdgcn_wmma_f32_16x16x32_f16(false, afrag, false, bfrag,
                                                 (short)0, acc, false, false);
  }

#pragma unroll
  for (int r = 0; r < 8; ++r) {
    int m = rowBase + hi * 8 + r;
    int n = colBase + lo;
    float v = acc[r];
    if (HAS_BIAS)  v += bias[n];
    if (DO_RELU)   v = fmaxf(v, 0.f);
    if (HAS_RESID) v += resid[(size_t)m * Ndim + n];
    C[(size_t)m * Ndim + n] = v;
  }
  (void)Mdim;
}

// ============================================================
// 5) Attention over BINS=5 tokens per (cell,batch,head); Dh=32
// ============================================================
__global__ void attn_kernel(const float* __restrict__ q, const float* __restrict__ k,
                            const float* __restrict__ v, float* __restrict__ o) {
  int blk = blockIdx.x;                  // ((c*NB)+b)*NH + h
  int h = blk % NH;
  int cb = blk / NH;
  int b = cb % NB, c = cb / NB;
  __shared__ float sq[NBINS][32], sk[NBINS][32], sv[NBINS][32], sa[NBINS][NBINS];
  int t = threadIdx.x;                   // 160 threads
  int s = t / 32, d = t % 32;
  int tok = (c * NBINS + s) * NB + b;
  int f = h * 32 + d;
  sq[s][d] = q[(size_t)tok * FF + f];
  sk[s][d] = k[(size_t)tok * FF + f];
  sv[s][d] = v[(size_t)tok * FF + f];
  __syncthreads();
  if (t < NBINS * NBINS) {
    int si = t / NBINS, ti = t % NBINS;
    float a = 0.f;
    for (int dd = 0; dd < 32; ++dd) a += sq[si][dd] * sk[ti][dd];
    sa[si][ti] = a * 0.17677669529663687f;   // 1/sqrt(32)
  }
  __syncthreads();
  if (t < NBINS) {
    float mxv = sa[t][0];
    for (int j = 1; j < NBINS; ++j) mxv = fmaxf(mxv, sa[t][j]);
    float sum = 0.f;
    for (int j = 0; j < NBINS; ++j) { float e = __expf(sa[t][j] - mxv); sa[t][j] = e; sum += e; }
    float inv = 1.0f / sum;
    for (int j = 0; j < NBINS; ++j) sa[t][j] *= inv;
  }
  __syncthreads();
  float a = 0.f;
  for (int j = 0; j < NBINS; ++j) a += sa[s][j] * sv[j][d];
  o[(size_t)tok * FF + f] = a;
}

// ============================================================
// 6) LayerNorm over F=256 per token
// ============================================================
__global__ void ln_kernel(const float* __restrict__ in, const float* __restrict__ g,
                          const float* __restrict__ bta, float* __restrict__ out) {
  int tok = blockIdx.x;
  int f = threadIdx.x;  // 256
  float v = in[(size_t)tok * FF + f];
  __shared__ float red[256];
  red[f] = v;
  __syncthreads();
  for (int s = 128; s > 0; s >>= 1) { if (f < s) red[f] += red[f + s]; __syncthreads(); }
  float mean = red[0] / (float)FF;
  __syncthreads();
  float dv = v - mean;
  red[f] = dv * dv;
  __syncthreads();
  for (int s = 128; s > 0; s >>= 1) { if (f < s) red[f] += red[f + s]; __syncthreads(); }
  float var = red[0] / (float)FF;
  out[(size_t)tok * FF + f] = dv * rsqrtf(var + 1e-5f) * g[f] + bta[f];
}

// ============================================================
// 7) Final adaptive max pool -> (5,4,32,32)
// ============================================================
__global__ void final_pool_kernel(const float* __restrict__ x, float* __restrict__ out) {
  int i = blockIdx.x * blockDim.x + threadIdx.x;
  if (i >= NBINS * NB * 32 * 32) return;
  int j  = i % 32;
  int ii = (i / 32) % 32;
  int b  = (i / (32 * 32)) % NB;
  int s  = i / (32 * 32 * NB);
  float mx = -3.4e38f;
  for (int e = 0; e < 2; ++e) {
    int c = j * 2 + e;
    int tok = (c * NBINS + s) * NB + b;
    for (int a = 0; a < 8; ++a)
      mx = fmaxf(mx, x[(size_t)tok * FF + ii * 8 + a]);
  }
  out[i] = mx;
}

// ============================================================
// host-side orchestration
// ============================================================
extern "C" void kernel_launch(void* const* d_in, const int* in_sizes, int n_in,
                              void* d_out, int out_size, void* d_ws, size_t ws_size,
                              hipStream_t stream) {
  const float* pos   = (const float*)d_in[0];
  const float* fin   = (const float*)d_in[1];
  const float* kp    = (const float*)d_in[2];
  const float* kpW   = (const float*)d_in[3];
  const float* bn_g  = (const float*)d_in[4];
  const float* bn_b  = (const float*)d_in[5];
  const float* Wemb  = (const float*)d_in[6];
  const float* Wq    = (const float*)d_in[7];
  const float* bq    = (const float*)d_in[8];
  const float* Wk    = (const float*)d_in[9];
  const float* bk    = (const float*)d_in[10];
  const float* Wv    = (const float*)d_in[11];
  const float* bv    = (const float*)d_in[12];
  const float* Wo    = (const float*)d_in[13];
  const float* bo    = (const float*)d_in[14];
  const float* ln1g  = (const float*)d_in[15];
  const float* ln1b  = (const float*)d_in[16];
  const float* W1f   = (const float*)d_in[17];
  const float* b1f   = (const float*)d_in[18];
  const float* W2f   = (const float*)d_in[19];
  const float* b2f   = (const float*)d_in[20];
  const float* ln2g  = (const float*)d_in[21];
  const float* ln2b  = (const float*)d_in[22];
  (void)in_sizes; (void)n_in; (void)out_size; (void)ws_size;

  const int NTOK = NCELL * NBINS * NB;   // 1280

  char* base = (char*)d_ws;
  size_t off = 0;
  auto carve = [&](size_t floats) {
    float* p = (float*)(base + off);
    off += ((floats * sizeof(float)) + 255) & ~(size_t)255;
    return p;
  };
  float* feat   = carve((size_t)NB * PN * CKP);
  float* stats  = carve(2 * CKP);
  int*   cellid = (int*)carve((size_t)NB * PN);
  float* Aemb   = carve((size_t)NTOK * DD);
  float* xbuf   = carve((size_t)NTOK * FF);
  float* qbuf   = carve((size_t)NTOK * FF);
  float* kbuf   = carve((size_t)NTOK * FF);   // also reused as FFN hidden
  float* vbuf   = carve((size_t)NTOK * FF);
  float* obuf   = carve((size_t)NTOK * FF);

  kpconv_kernel<<<(NB * PN + 255) / 256, 256, 0, stream>>>(pos, fin, kp, kpW, feat);
  bn_stats_kernel<<<CKP, 256, 0, stream>>>(feat, stats);
  bn_apply_kernel<<<(NB * PN * CKP + 255) / 256, 256, 0, stream>>>(feat, stats, bn_g, bn_b);
  cellid_kernel<<<(NB * PN + 255) / 256, 256, 0, stream>>>(pos, cellid);
  cluster_pool_kernel<<<NCELL * NB, DD, 0, stream>>>(pos, feat, cellid, Aemb);

  dim3 gF(FF / 16, NTOK / 64);
  dim3 gH(HIDN / 16, NTOK / 64);

  // embed: (1280x64) @ (64x256)
  gemm_wmma_kernel<DD, false, false, false><<<gF, 128, 0, stream>>>(
      Aemb, Wemb, nullptr, nullptr, xbuf, NTOK, FF);

  for (int l = 0; l < NL; ++l) {
    gemm_wmma_kernel<FF, true, false, false><<<gF, 128, 0, stream>>>(
        xbuf, Wq + (size_t)l * FF * FF, bq + l * FF, nullptr, qbuf, NTOK, FF);
    gemm_wmma_kernel<FF, true, false, false><<<gF, 128, 0, stream>>>(
        xbuf, Wk + (size_t)l * FF * FF, bk + l * FF, nullptr, kbuf, NTOK, FF);
    gemm_wmma_kernel<FF, true, false, false><<<gF, 128, 0, stream>>>(
        xbuf, Wv + (size_t)l * FF * FF, bv + l * FF, nullptr, vbuf, NTOK, FF);
    attn_kernel<<<NCELL * NB * NH, NBINS * 32, 0, stream>>>(qbuf, kbuf, vbuf, obuf);
    gemm_wmma_kernel<FF, true, false, true><<<gF, 128, 0, stream>>>(
        obuf, Wo + (size_t)l * FF * FF, bo + l * FF, xbuf, qbuf, NTOK, FF);
    ln_kernel<<<NTOK, FF, 0, stream>>>(qbuf, ln1g + l * FF, ln1b + l * FF, xbuf);
    gemm_wmma_kernel<FF, true, true, false><<<gH, 128, 0, stream>>>(
        xbuf, W1f + (size_t)l * FF * HIDN, b1f + l * HIDN, nullptr, kbuf, NTOK, HIDN);
    gemm_wmma_kernel<HIDN, true, false, true><<<gF, 128, 0, stream>>>(
        kbuf, W2f + (size_t)l * HIDN * FF, b2f + l * FF, xbuf, qbuf, NTOK, FF);
    ln_kernel<<<NTOK, FF, 0, stream>>>(qbuf, ln2g + l * FF, ln2b + l * FF, xbuf);
  }

  final_pool_kernel<<<(NBINS * NB * 32 * 32 + 255) / 256, 256, 0, stream>>>(xbuf, (float*)d_out);
}
